// sLSTMNetwork_16724602650753
// MI455X (gfx1250) — matrix operand order repaired and checked by reading, not compiled, not run on hardware
//
#include <hip/hip_runtime.h>
#include <hip/hip_bf16.h>

// ---------------------------------------------------------------------------
// sLSTM (L=2, H=512, T=512, B=64) for gfx1250.
// GEMMs via v_wmma_f32_16x16x32_f16 (f16 operands, f32 accumulate); all
// elementwise/reduction work fused into one cell kernel per layer-step.
// ---------------------------------------------------------------------------

#define L_   2
#define H_   512
#define H4_  2048
#define B_   64
#define T_   512

typedef __attribute__((ext_vector_type(16))) _Float16 v16h;
typedef __attribute__((ext_vector_type(8)))  float    v8f;
typedef __attribute__((ext_vector_type(4)))  float    v4f;

// ---------------------------------------------------------------------------
// Weight fp32 -> fp16 conversion (once per launch; deterministic).
// ---------------------------------------------------------------------------
__global__ void convert_weights(const float* __restrict__ Wx,
                                const float* __restrict__ Wh,
                                _Float16* __restrict__ Wxh,
                                _Float16* __restrict__ Whh, int n) {
  int i = blockIdx.x * blockDim.x + threadIdx.x;
  int stride = gridDim.x * blockDim.x;
  for (; i < n; i += stride) {
    Wxh[i] = (_Float16)Wx[i];
    Whh[i] = (_Float16)Wh[i];
  }
}

// ---------------------------------------------------------------------------
// GEMM: Out[64,2048] = A[64,512] @ W^T  (W row-major [2048,512], f16).
// blockIdx.z selects {A0,W0,O0} (input path) or {A1,W1,O1} (recurrent path).
// One 16x16 output tile per wave; K-loop of 16 WMMA ops.
// Operand layouts: contiguous k = 16*hf + s in A and B alike, so plain
// coalesced global loads produce correct fragments (no LDS needed).
// ---------------------------------------------------------------------------
__global__ __launch_bounds__(256)
void slstm_gemm(const float* __restrict__ A0, const _Float16* __restrict__ W0,
                float* __restrict__ O0,
                const float* __restrict__ A1, const _Float16* __restrict__ W1,
                float* __restrict__ O1) {
  const int mat = blockIdx.z;
  const float*    A   = mat ? A1 : A0;
  const _Float16* W   = mat ? W1 : W0;
  float*          Out = mat ? O1 : O0;

  const int wave = threadIdx.x >> 5;
  const int lane = threadIdx.x & 31;
  const int hf   = lane >> 4;     // half-wave select
  const int l15  = lane & 15;

  const int n0 = (blockIdx.x * 8 + wave) * 16;  // 0..2032
  const int m0 = blockIdx.y * 16;               // 0..48

  const float*    Arow = A + (size_t)(m0 + l15) * H_;
  const _Float16* Wrow = W + (size_t)(n0 + l15) * H_;

  v8f acc = {};
  for (int k0 = 0; k0 < H_; k0 += 32) {
    // A fragment: lane<16 -> K k0..k0+15, lane>=16 -> +16.
    const int ka = k0 + hf * 16;
    v4f a0 = *(const v4f*)(Arow + ka);
    v4f a1 = *(const v4f*)(Arow + ka + 4);
    v4f a2 = *(const v4f*)(Arow + ka + 8);
    v4f a3 = *(const v4f*)(Arow + ka + 12);
    // B fragment: B[k][n] = W[n][k]; lane<16 -> K k0..k0+15, lane>=16 -> +16.
    const int kb = k0 + hf * 16;
    v16h bf = *(const v16h*)(Wrow + kb);
    if (k0 + 32 < H_) __builtin_prefetch(Wrow + kb + 32, 0, 0);

    v16h af;
    af[0]  = (_Float16)a0[0];  af[1]  = (_Float16)a0[1];
    af[2]  = (_Float16)a0[2];  af[3]  = (_Float16)a0[3];
    af[4]  = (_Float16)a1[0];  af[5]  = (_Float16)a1[1];
    af[6]  = (_Float16)a1[2];  af[7]  = (_Float16)a1[3];
    af[8]  = (_Float16)a2[0];  af[9]  = (_Float16)a2[1];
    af[10] = (_Float16)a2[2];  af[11] = (_Float16)a2[3];
    af[12] = (_Float16)a3[0];  af[13] = (_Float16)a3[1];
    af[14] = (_Float16)a3[2];  af[15] = (_Float16)a3[3];

    acc = __builtin_amdgcn_wmma_f32_16x16x32_f16(
        false, af, false, bf, (short)0, acc, false, false);
  }

  // D layout: lane&15 = N, VGPR j = row (m0 + 8*hf + j).
  float* Ocol = Out + (size_t)(m0 + hf * 8) * H4_ + n0 + l15;
#pragma unroll
  for (int j = 0; j < 8; ++j) Ocol[(size_t)j * H4_] = acc[j];
}

// ---------------------------------------------------------------------------
// Block-wide sum over 256 threads (LDS tree).
// ---------------------------------------------------------------------------
__device__ __forceinline__ float block_sum(float v, float* red) {
  const int tid = threadIdx.x;
  red[tid] = v;
  __syncthreads();
#pragma unroll
  for (int s = 128; s > 0; s >>= 1) {
    if (tid < s) red[tid] += red[tid + s];
    __syncthreads();
  }
  float r = red[0];
  __syncthreads();
  return r;
}

__device__ __forceinline__ float clamp5(float x) {
  return fminf(fmaxf(x, -5.0f), 5.0f);
}

// ---------------------------------------------------------------------------
// Fused sLSTM cell: LN(S+bx), LN(U+bh), sum, normalized-exp i/f gates,
// cell update + LN, output gate + two LNs. One block per batch row.
// ---------------------------------------------------------------------------
__global__ __launch_bounds__(256)
void slstm_cell(const float* __restrict__ S, const float* __restrict__ U,
                const float* __restrict__ bx, const float* __restrict__ bh,
                const float* __restrict__ gin, const float* __restrict__ bin,
                const float* __restrict__ ghu, const float* __restrict__ bhu,
                const float* __restrict__ gc,  const float* __restrict__ bc,
                const float* __restrict__ ghn, const float* __restrict__ bhn,
                const float* __restrict__ go,  const float* __restrict__ bo,
                float* __restrict__ c_state, float* __restrict__ h_state) {
  __shared__ float z[H4_];
  __shared__ float red[256];
  const int b   = blockIdx.x;
  const int tid = threadIdx.x;

  // ---- dual LayerNorm over 4H = 2048 ----
  float z1[8], z2[8];
  float s1 = 0.f, s2 = 0.f;
#pragma unroll
  for (int r = 0; r < 8; ++r) {
    int j = tid + r * 256;
    z1[r] = S[(size_t)b * H4_ + j] + bx[j];
    z2[r] = U[(size_t)b * H4_ + j] + bh[j];
    s1 += z1[r];
    s2 += z2[r];
  }
  const float m1 = block_sum(s1, red) * (1.0f / H4_);
  const float m2 = block_sum(s2, red) * (1.0f / H4_);
  float v1 = 0.f, v2 = 0.f;
#pragma unroll
  for (int r = 0; r < 8; ++r) {
    float d1 = z1[r] - m1; v1 += d1 * d1;
    float d2 = z2[r] - m2; v2 += d2 * d2;
  }
  const float rs1 = rsqrtf(block_sum(v1, red) * (1.0f / H4_) + 1e-5f);
  const float rs2 = rsqrtf(block_sum(v2, red) * (1.0f / H4_) + 1e-5f);
#pragma unroll
  for (int r = 0; r < 8; ++r) {
    int j = tid + r * 256;
    float gx = (z1[r] - m1) * rs1 * gin[j] + bin[j];
    float gv = (z2[r] - m2) * rs2 * ghu[j] + bhu[j];
    z[j] = gx + gv;
  }
  __syncthreads();

  // ---- gates: i, f, g, o each of size H=512; 2 elements per thread ----
  float iv[2], fv[2], gg[2], ov[2];
  float si = 0.f, sf = 0.f;
#pragma unroll
  for (int r = 0; r < 2; ++r) {
    int hh = tid + r * 256;
    iv[r] = z[hh];
    fv[r] = z[H_ + hh];
    gg[r] = z[2 * H_ + hh];
    ov[r] = z[3 * H_ + hh];
    si += iv[r];
    sf += fv[r];
  }
  const float mi = block_sum(si, red) * (1.0f / H_);
  const float mf = block_sum(sf, red) * (1.0f / H_);
  float ei[2], ef[2];
  float sei = 0.f, sef = 0.f;
#pragma unroll
  for (int r = 0; r < 2; ++r) {
    ei[r] = expf(clamp5(iv[r] - mi)); sei += ei[r];
    ef[r] = expf(clamp5(fv[r] - mf)); sef += ef[r];
  }
  const float di = 1.0f / (block_sum(sei, red) + 1e-6f);
  const float df = 1.0f / (block_sum(sef, red) + 1e-6f);

  // ---- cell update + LN over H ----
  float cn[2];
  float sc = 0.f;
#pragma unroll
  for (int r = 0; r < 2; ++r) {
    int hh = tid + r * 256;
    cn[r] = ef[r] * df * c_state[(size_t)b * H_ + hh] +
            ei[r] * di * tanhf(gg[r]);
    sc += cn[r];
  }
  const float mc = block_sum(sc, red) * (1.0f / H_);
  float vc = 0.f;
#pragma unroll
  for (int r = 0; r < 2; ++r) { float d = cn[r] - mc; vc += d * d; }
  const float rsc = rsqrtf(block_sum(vc, red) * (1.0f / H_) + 1e-5f);

  float hl[2];
  float sh = 0.f;
#pragma unroll
  for (int r = 0; r < 2; ++r) {
    int hh = tid + r * 256;
    float cl = (cn[r] - mc) * rsc * gc[hh] + bc[hh];
    c_state[(size_t)b * H_ + hh] = cl;           // carry is the LN'd cell
    float sig = 1.0f / (1.0f + expf(-ov[r]));
    hl[r] = sig * tanhf(cl);
    sh += hl[r];
  }
  const float mh = block_sum(sh, red) * (1.0f / H_);
  float vh = 0.f;
#pragma unroll
  for (int r = 0; r < 2; ++r) { float d = hl[r] - mh; vh += d * d; }
  const float rsh = rsqrtf(block_sum(vh, red) * (1.0f / H_) + 1e-5f);

  float h2[2];
  float sh2 = 0.f;
#pragma unroll
  for (int r = 0; r < 2; ++r) {
    int hh = tid + r * 256;
    h2[r] = (hl[r] - mh) * rsh * ghn[hh] + bhn[hh];
    sh2 += h2[r];
  }
  const float mo = block_sum(sh2, red) * (1.0f / H_);
  float vo = 0.f;
#pragma unroll
  for (int r = 0; r < 2; ++r) { float d = h2[r] - mo; vo += d * d; }
  const float rso = rsqrtf(block_sum(vo, red) * (1.0f / H_) + 1e-5f);
#pragma unroll
  for (int r = 0; r < 2; ++r) {
    int hh = tid + r * 256;
    h_state[(size_t)b * H_ + hh] = (h2[r] - mo) * rso * go[hh] + bo[hh];
  }
}

// ---------------------------------------------------------------------------
// Final FC on last layer's h (O=1) and state copy into d_out.
// ---------------------------------------------------------------------------
__global__ __launch_bounds__(256)
void slstm_fc(const float* __restrict__ hlast, const float* __restrict__ fcw,
              const float* __restrict__ fcb, float* __restrict__ out) {
  __shared__ float red[256];
  const int b = blockIdx.x, tid = threadIdx.x;
  float s = 0.f;
#pragma unroll
  for (int r = 0; r < 2; ++r) {
    int hh = tid + r * 256;
    s += hlast[(size_t)b * H_ + hh] * fcw[hh];
  }
  s = block_sum(s, red);
  if (tid == 0) out[b] = s + fcb[0];
}

__global__ void copy_states(const float* __restrict__ hs,
                            const float* __restrict__ cs,
                            float* __restrict__ out) {
  int i = blockIdx.x * blockDim.x + threadIdx.x;
  if (i < L_ * B_ * H_) {
    out[B_ + i] = hs[i];
    out[B_ + L_ * B_ * H_ + i] = cs[i];
  }
}

// ---------------------------------------------------------------------------
// Launch: init states, convert weights, 512 x (2 x (GEMM, cell)), FC + copy.
// ---------------------------------------------------------------------------
extern "C" void kernel_launch(void* const* d_in, const int* in_sizes, int n_in,
                              void* d_out, int out_size, void* d_ws,
                              size_t ws_size, hipStream_t stream) {
  const float* x   = (const float*)d_in[0];
  const float* h0  = (const float*)d_in[1];
  const float* c0  = (const float*)d_in[2];
  const float* Wx  = (const float*)d_in[3];
  const float* bx  = (const float*)d_in[4];
  const float* Wh  = (const float*)d_in[5];
  const float* bh  = (const float*)d_in[6];
  const float* gin = (const float*)d_in[7];
  const float* bin = (const float*)d_in[8];
  const float* ghu = (const float*)d_in[9];
  const float* bhu = (const float*)d_in[10];
  const float* gc  = (const float*)d_in[11];
  const float* bc  = (const float*)d_in[12];
  const float* ghn = (const float*)d_in[13];
  const float* bhn = (const float*)d_in[14];
  const float* go  = (const float*)d_in[15];
  const float* bo  = (const float*)d_in[16];
  const float* fcw = (const float*)d_in[17];
  const float* fcb = (const float*)d_in[18];
  float* out = (float*)d_out;

  // Workspace layout (bytes):
  //   S        : 64*2048 f32          @ 0        (512 KB)
  //   U        : 64*2048 f32          @ 524288   (512 KB)
  //   h_state  : L*B*H f32            @ 1048576  (256 KB)
  //   c_state  : L*B*H f32            @ 1310720  (256 KB)
  //   Wx (f16) : L*2048*512           @ 1572864  (4 MB)
  //   Wh (f16) : L*2048*512           @ 5767168  (4 MB)
  char* ws = (char*)d_ws;
  float*    S   = (float*)(ws);
  float*    U   = (float*)(ws + 524288);
  float*    hs  = (float*)(ws + 1048576);
  float*    cs  = (float*)(ws + 1310720);
  _Float16* Wxh = (_Float16*)(ws + 1572864);
  _Float16* Whh = (_Float16*)(ws + 5767168);

  const size_t state_bytes = (size_t)L_ * B_ * H_ * sizeof(float);
  hipMemcpyAsync(hs, h0, state_bytes, hipMemcpyDeviceToDevice, stream);
  hipMemcpyAsync(cs, c0, state_bytes, hipMemcpyDeviceToDevice, stream);
  convert_weights<<<512, 256, 0, stream>>>(Wx, Wh, Wxh, Whh,
                                           L_ * H4_ * H_);

  for (int t = 0; t < T_; ++t) {
    for (int l = 0; l < L_; ++l) {
      const float* inp =
          (l == 0) ? (x + (size_t)t * B_ * H_) : (hs + (size_t)(l - 1) * B_ * H_);
      slstm_gemm<<<dim3(16, 4, 2), 256, 0, stream>>>(
          inp, Wxh + (size_t)l * H4_ * H_, S,
          hs + (size_t)l * B_ * H_, Whh + (size_t)l * H4_ * H_, U);
      slstm_cell<<<B_, 256, 0, stream>>>(
          S, U, bx + l * H4_, bh + l * H4_,
          gin + l * H4_, bin + l * H4_, ghu + l * H4_, bhu + l * H4_,
          gc + l * H_, bc + l * H_, ghn + l * H_, bhn + l * H_,
          go + l * H_, bo + l * H_,
          cs + (size_t)l * B_ * H_, hs + (size_t)l * B_ * H_);
    }
  }

  slstm_fc<<<B_, 256, 0, stream>>>(hs + (size_t)(L_ - 1) * B_ * H_, fcw, fcb,
                                   out);
  copy_states<<<256, 256, 0, stream>>>(hs, cs, out);
}
